// AdditiveAttention_7619271983272
// MI455X (gfx1250) — matrix operand order repaired and checked by reading, not compiled, not run on hardware
//
#include <hip/hip_runtime.h>
#include <hip/hip_bf16.h>

typedef __attribute__((ext_vector_type(16))) _Float16 v16h;
typedef __attribute__((ext_vector_type(8)))  float    v8f;

#define Bn 4
#define Qn 256
#define Kn 1024
#define Dn 512    // DQ = DK = DV
#define Hn 128

__device__ __forceinline__ float fast_tanh(float x) {
    // tanh(x) = 1 - 2/(exp(2x)+1); lowers to v_exp_f32 + rcp path.
    float e = __expf(2.0f * x);
    return 1.0f - 2.0f / (e + 1.0f);
}

// ---------------------------------------------------------------------------
// f32 -> f16 copy (used once for w_q and w_k; 64K elements each)
// ---------------------------------------------------------------------------
__global__ __launch_bounds__(256) void cvt16_kernel(const float* __restrict__ src,
                                                    _Float16* __restrict__ dst) {
    const size_t i = (size_t)blockIdx.x * 256 + threadIdx.x;
    dst[i] = (_Float16)src[i];
}

// ---------------------------------------------------------------------------
// OUT = X[rows x 512] * Wh[128 x 512]^T, f32 accumulate via WMMA f16.
// One wave per 16-row tile; 8 n-tiles of 16 cols; K reduced 32 at a time.
// TRANS=false: OUT[row][h] (row-major, ld = 128)
// TRANS=true : OUT[h][row] (h-major, ld = ldT)  -- for kh so the scores pass
//              reads coalesced
// ---------------------------------------------------------------------------
template <bool TRANS>
__global__ __launch_bounds__(128) void proj_wmma(const float* __restrict__ X,
                                                 const _Float16* __restrict__ Wh,
                                                 float* __restrict__ OUT,
                                                 int ldT) {
    const int lane  = threadIdx.x & 31;
    const int wave  = threadIdx.x >> 5;
    const int m0    = (blockIdx.x * 4 + wave) * 16;
    const int lrow  = lane & 15;
    const int hi    = lane >> 4;

    v8f acc[8] = {};

    const float* xrow = X + (size_t)(m0 + lrow) * Dn;

    for (int kk = 0; kk < Dn / 32; ++kk) {
        const int k0 = kk * 32;
        // Prefetch the next A chunk of this row (speculative, dropped at end).
        __builtin_prefetch(xrow + k0 + 32, 0, 1);
        // A tile (16x32 f16): lane<16 holds K {0..7,16..23}, lane>=16 {8..15,24..31}
        v16h a;
        const float* p0 = xrow + k0 + hi * 8;
        const float* p1 = xrow + k0 + 16 + hi * 8;
#pragma unroll
        for (int i = 0; i < 8; ++i) {
            a[i]     = (_Float16)p0[i];
            a[i + 8] = (_Float16)p1[i];
        }
#pragma unroll
        for (int n = 0; n < 8; ++n) {
            // B tile (32x16 f16): lane holds column n=lrow, K {0..7,16..23} / {8..15,24..31}.
            const _Float16* wrow = Wh + (size_t)(n * 16 + lrow) * Dn + k0 + hi * 8;
            v16h bt;
#pragma unroll
            for (int i = 0; i < 8; ++i) { bt[i] = wrow[i]; bt[i + 8] = wrow[16 + i]; }
            acc[n] = __builtin_amdgcn_wmma_f32_16x16x32_f16(
                false, a, false, bt, (short)0, acc[n], false, false);
        }
    }
#pragma unroll
    for (int n = 0; n < 8; ++n)
#pragma unroll
        for (int r = 0; r < 8; ++r) {
            const int row = m0 + r + hi * 8;
            const int h   = n * 16 + lrow;
            if (TRANS)
                OUT[(size_t)h * ldT + row] = acc[n][r];
            else
                OUT[(size_t)row * Hn + h] = acc[n][r];
        }
}

// ---------------------------------------------------------------------------
// scores[b,q,k] = sum_h w_v[h]*tanh(qh[b,q,h]+kh_t[h][b*K+k]); masked.
// kh_t is [H][B*K] so consecutive lanes read consecutive k (coalesced b32).
// ---------------------------------------------------------------------------
__global__ __launch_bounds__(256) void scores_kernel(const float* __restrict__ qh,
                                                     const float* __restrict__ kh_t,
                                                     const float* __restrict__ wv,
                                                     const int* __restrict__ vlen,
                                                     float* __restrict__ scores) {
    __shared__ float qs[Hn];
    __shared__ float ws[Hn];
    const int bq = blockIdx.y;               // b*Q + q
    const int b  = bq / Qn;
    const int k  = blockIdx.x * 256 + threadIdx.x;
    if (threadIdx.x < Hn) {
        qs[threadIdx.x] = qh[(size_t)bq * Hn + threadIdx.x];
        ws[threadIdx.x] = wv[threadIdx.x];
    }
    __syncthreads();
    const float* kcol = kh_t + (size_t)b * Kn + k;   // stride Bn*Kn per h
    float acc = 0.0f;
#pragma unroll 8
    for (int h = 0; h < Hn; ++h)
        acc += ws[h] * fast_tanh(qs[h] + kcol[(size_t)h * (Bn * Kn)]);
    const int vl = vlen[b];
    scores[(size_t)bq * Kn + k] = (k < vl) ? acc : -1.0e6f;
}

// ---------------------------------------------------------------------------
// Row softmax over K=1024, emit f16 attention weights.
// ---------------------------------------------------------------------------
__global__ __launch_bounds__(256) void softmax_kernel(const float* __restrict__ scores,
                                                      _Float16* __restrict__ attn) {
    __shared__ float red[256];
    const int bq = blockIdx.x;
    const int t  = threadIdx.x;
    const float* row = scores + (size_t)bq * Kn;
    float x[4];
    float m = -3.4e38f;
#pragma unroll
    for (int i = 0; i < 4; ++i) { x[i] = row[t + i * 256]; m = fmaxf(m, x[i]); }
    red[t] = m; __syncthreads();
    for (int s = 128; s > 0; s >>= 1) {
        if (t < s) red[t] = fmaxf(red[t], red[t + s]);
        __syncthreads();
    }
    m = red[0]; __syncthreads();
    float sum = 0.0f;
#pragma unroll
    for (int i = 0; i < 4; ++i) { x[i] = __expf(x[i] - m); sum += x[i]; }
    red[t] = sum; __syncthreads();
    for (int s = 128; s > 0; s >>= 1) {
        if (t < s) red[t] += red[t + s];
        __syncthreads();
    }
    const float inv = 1.0f / red[0];
#pragma unroll
    for (int i = 0; i < 4; ++i)
        attn[(size_t)bq * Kn + t + i * 256] = (_Float16)(x[i] * inv);
}

// ---------------------------------------------------------------------------
// v[B,K,DV] f32 -> vt[B,DV,K] f16 (K-major so GEMM B tiles are contiguous)
// ---------------------------------------------------------------------------
__global__ __launch_bounds__(256) void vt_kernel(const float* __restrict__ v,
                                                 _Float16* __restrict__ vt) {
    const size_t idx = (size_t)blockIdx.x * 256 + threadIdx.x;  // over B*Dn*Kn
    const int    k   = (int)(idx % Kn);
    const size_t r   = idx / Kn;
    const int    d   = (int)(r % Dn);
    const int    b   = (int)(r / Dn);
    vt[idx] = (_Float16)v[((size_t)b * Kn + k) * Dn + d];
}

// ---------------------------------------------------------------------------
// out[b,q,:] = attn[b,Q,K] (f16) x vt[b,DV,K]^T via WMMA.
// One wave per 16x64 output tile; 32 k-steps x 4 WMMAs.
// ---------------------------------------------------------------------------
__global__ __launch_bounds__(128) void out_wmma(const _Float16* __restrict__ attn,
                                                const _Float16* __restrict__ vt,
                                                float* __restrict__ out) {
    const int lane = threadIdx.x & 31;
    const int wave = threadIdx.x >> 5;
    const int job  = blockIdx.x * 4 + wave;     // 512 jobs
    const int b    = job >> 7;                  // / (16*8)
    const int rem  = job & 127;
    const int m0   = (rem >> 3) * 16;
    const int n0   = (rem & 7) * 64;
    const int lrow = lane & 15;
    const int hi   = lane >> 4;

    v8f acc[4] = {};

    const _Float16* arow  = attn + ((size_t)b * Qn + m0 + lrow) * Kn;
    const _Float16* vbase = vt + (size_t)b * Dn * Kn;

    for (int kk = 0; kk < Kn / 32; ++kk) {
        const int k0 = kk * 32;
        __builtin_prefetch(arow + k0 + 32, 0, 1);
        v16h a;
        const _Float16* p0 = arow + k0 + hi * 8;
        const _Float16* p1 = arow + k0 + 16 + hi * 8;
#pragma unroll
        for (int i = 0; i < 8; ++i) { a[i] = p0[i]; a[i + 8] = p1[i]; }
#pragma unroll
        for (int n = 0; n < 4; ++n) {
            const _Float16* vrow = vbase + (size_t)(n0 + n * 16 + lrow) * Kn + k0 + hi * 8;
            v16h bt;
#pragma unroll
            for (int i = 0; i < 8; ++i) { bt[i] = vrow[i]; bt[i + 8] = vrow[16 + i]; }
            acc[n] = __builtin_amdgcn_wmma_f32_16x16x32_f16(
                false, a, false, bt, (short)0, acc[n], false, false);
        }
    }
#pragma unroll
    for (int n = 0; n < 4; ++n)
#pragma unroll
        for (int r = 0; r < 8; ++r)
            out[((size_t)b * Qn + m0 + r + hi * 8) * Dn + n0 + n * 16 + lrow] = acc[n][r];
}

extern "C" void kernel_launch(void* const* d_in, const int* in_sizes, int n_in,
                              void* d_out, int out_size, void* d_ws, size_t ws_size,
                              hipStream_t stream) {
    const float* q  = (const float*)d_in[0];
    const float* k  = (const float*)d_in[1];
    const float* v  = (const float*)d_in[2];
    const int*   vl = (const int*)d_in[3];
    const float* wq = (const float*)d_in[4];
    const float* wk = (const float*)d_in[5];
    const float* wv = (const float*)d_in[6];
    float* out = (float*)d_out;

    char* ws = (char*)d_ws;
    float*    qh   = (float*)(ws);                              // 4*256*128*4   = 512 KB
    float*    khT  = (float*)(ws + (size_t)524288);             // [128][4096]f32= 2 MB
    float*    sc   = (float*)(ws + (size_t)2621440);            // 4*256*1024*4  = 4 MB
    _Float16* attn = (_Float16*)(ws + (size_t)6815744);         // 4*256*1024*2  = 2 MB
    _Float16* vt   = (_Float16*)(ws + (size_t)8912896);         // 4*512*1024*2  = 4 MB
    _Float16* wqh  = (_Float16*)(ws + (size_t)13107200);        // 128*512*2     = 128 KB
    _Float16* wkh  = (_Float16*)(ws + (size_t)13238272);        // 128*512*2     = 128 KB

    // One-time f16 conversion of the shared projection weights.
    cvt16_kernel<<<(Hn * Dn) / 256, 256, 0, stream>>>(wq, wqh);
    cvt16_kernel<<<(Hn * Dn) / 256, 256, 0, stream>>>(wk, wkh);
    // Projections: qh = q*wq^T (row-major), khT = (k*wk^T)^T ([H][B*K])
    proj_wmma<false><<<(Bn * Qn) / 64, 128, 0, stream>>>(q, wqh, qh, 0);
    proj_wmma<true ><<<(Bn * Kn) / 64, 128, 0, stream>>>(k, wkh, khT, Bn * Kn);
    // v transpose to f16 (K-major)
    vt_kernel<<<(Bn * Dn * Kn) / 256, 256, 0, stream>>>(v, vt);
    // Additive scores with tanh + mask (coalesced khT reads)
    scores_kernel<<<dim3(Kn / 256, Bn * Qn), 256, 0, stream>>>(qh, khT, wv, vl, sc);
    // Softmax -> f16 attention
    softmax_kernel<<<Bn * Qn, 256, 0, stream>>>(sc, attn);
    // out = attn @ v
    out_wmma<<<(Bn * (Qn / 16) * (Dn / 64)) / 4, 128, 0, stream>>>(attn, vt, out);
}
